// RetentionLayer_9079560864066
// MI455X (gfx1250) — matrix operand order repaired and checked by reading, not compiled, not run on hardware
//
#include <hip/hip_runtime.h>
#include <hip/hip_bf16.h>

// ---------------------------------------------------------------------------
// RetentionLayer for MI455X (gfx1250, wave32, WMMA 16x16x32 f16,
// async-to-LDS staging + Tensor Data Mover)
// B=4, T=4096, D=1024, H=4, dk=256, dv=512, CHUNK=64, N=64
// ---------------------------------------------------------------------------

typedef _Float16 v16h __attribute__((ext_vector_type(16)));
typedef _Float16 v8h  __attribute__((ext_vector_type(8)));
typedef float    v8f  __attribute__((ext_vector_type(8)));
typedef unsigned int u32x4 __attribute__((ext_vector_type(4)));
typedef unsigned int u32x8 __attribute__((ext_vector_type(8)));

union Frag { v16h v; v8h h[2]; };

__device__ __forceinline__ v8f wmma16x16x32(v16h a, v16h b, v8f c) {
  // D = A(16x32 f16) * B(32x16 f16) + C(16x16 f32)
  return __builtin_amdgcn_wmma_f32_16x16x32_f16(
      /*neg_a=*/false, a, /*neg_b=*/false, b,
      /*c_mod=*/(short)0, c, /*reuse_a=*/false, /*reuse_b=*/false);
}

// generic -> LDS(addrspace 3) byte offset (32-bit)
__device__ __forceinline__ unsigned lds_off(const void* p) {
  return (unsigned)(unsigned long long)
      (const __attribute__((address_space(3))) void*)p;
}

// per-lane async copy of 16B global -> LDS (uses ASYNCcnt, no VGPR bounce)
__device__ __forceinline__ void async_b128(unsigned lds, const void* g) {
  asm volatile("global_load_async_to_lds_b128 %0, %1, off"
               :: "v"(lds), "v"(g) : "memory");
}
__device__ __forceinline__ void wait_async0() {
  asm volatile("s_wait_asynccnt 0x0" ::: "memory");
}

// Tensor Data Mover: 2D tile (rows x cols f16 elements, row stride in elems)
// global -> LDS, contiguous in LDS. Issue once per wave; tracked by TENSORcnt.
__device__ __forceinline__ void tdm_load_2d_f16(unsigned lds, const void* gp,
                                                unsigned rows, unsigned cols,
                                                unsigned stride_elems) {
  unsigned long long ga = (unsigned long long)gp;
  u32x4 g0 = { 1u,                                   // count=1 (valid user D#)
               lds,                                  // lds_addr (bytes)
               (unsigned)ga,                         // global_addr[31:0]
               ((unsigned)(ga >> 32) & 0x01FFFFFFu)  // global_addr[56:32]
                   | 0x80000000u };                  // type=2 ("image")
  u32x8 g1 = { 0x00010000u,        // wg_mask=0, data_size=1 (2B)
               cols << 16,         // atomic_barrier=0 | tensor_dim0 lo16
               rows << 16,         // tensor_dim0 hi16=0 | tensor_dim1 lo16
               cols << 16,         // tensor_dim1 hi16=0 | tile_dim0
               rows,               // tile_dim1 | tile_dim2=0
               stride_elems,       // tensor_dim0_stride lo32
               0u,                 // stride0 hi16=0 | stride1 lo16=0
               0u };               // stride1 hi32=0
  asm volatile("tensor_load_to_lds %0, %1" :: "s"(g0), "s"(g1) : "memory");
}

#define BB 4
#define TT 4096
#define DD 1024
#define HH 4
#define DKk 256
#define DVv 512
#define CC 64
#define NN 64
#define ROWS (BB*TT)   // 16384

// ---------------------------------------------------------------------------
// 1) LayerNorm -> f16
// ---------------------------------------------------------------------------
__global__ __launch_bounds__(256) void ln_f16(const float* __restrict__ x,
                                              const float* __restrict__ w,
                                              const float* __restrict__ b,
                                              _Float16* __restrict__ out) {
  int row = blockIdx.x;
  int tid = threadIdx.x;
  __shared__ float red[256];
  const float* xr = x + (size_t)row * DD;
  float vals[4];
  float s = 0.f, s2 = 0.f;
  for (int j = 0; j < 4; ++j) {
    float v = xr[tid + 256 * j];
    vals[j] = v; s += v; s2 += v * v;
  }
  red[tid] = s; __syncthreads();
  for (int off = 128; off; off >>= 1) {
    if (tid < off) red[tid] += red[tid + off];
    __syncthreads();
  }
  float mu = red[0] * (1.f / DD);
  __syncthreads();
  red[tid] = s2; __syncthreads();
  for (int off = 128; off; off >>= 1) {
    if (tid < off) red[tid] += red[tid + off];
    __syncthreads();
  }
  float var = red[0] * (1.f / DD) - mu * mu;
  float inv = rsqrtf(var + 1e-5f);
  _Float16* orow = out + (size_t)row * DD;
  for (int j = 0; j < 4; ++j) {
    int col = tid + 256 * j;
    orow[col] = (_Float16)((vals[j] - mu) * inv * w[col] + b[col]);
  }
}

// ---------------------------------------------------------------------------
// 2) Transpose+convert weight: src[K][N] f32 -> dst[N][K] f16
// ---------------------------------------------------------------------------
__global__ __launch_bounds__(256) void cvtT(const float* __restrict__ src,
                                            _Float16* __restrict__ dst,
                                            int K, int N) {
  size_t total = (size_t)K * N;
  for (size_t e = blockIdx.x * 256ull + threadIdx.x; e < total;
       e += (size_t)gridDim.x * 256ull) {
    size_t n = e / (size_t)K, k = e % (size_t)K;
    dst[e] = (_Float16)src[k * (size_t)N + n];
  }
}

// ---------------------------------------------------------------------------
// 3) GEMM: C[M][N] f32 = A[M][K] f16 * Bt[N][K] f16 (+ resid)
//    block tile 64x64, 8 waves, wave tile 16x32, K step 32 (WMMA f16)
//    Staging via global_load_async_to_lds_b128 (ASYNCcnt).
// ---------------------------------------------------------------------------
__global__ __launch_bounds__(256) void gemm_f16(const _Float16* __restrict__ A,
                                                const _Float16* __restrict__ Bt,
                                                const float* __restrict__ resid,
                                                float* __restrict__ Cout,
                                                int M, int N, int K) {
  __shared__ alignas(32) _Float16 As[64][32];
  __shared__ alignas(32) _Float16 Bs[64][32];
  int tid = threadIdx.x;
  int wave = tid >> 5, lane = tid & 31;
  int rl = lane & 15, hi = lane >> 4;
  int m0 = blockIdx.y * 64, n0 = blockIdx.x * 64;
  int mt = wave & 3;
  int ntb = (wave >> 2) * 2;  // this wave's n tiles: ntb, ntb+1
  v8f acc0 = {}, acc1 = {};
  int lrow = tid >> 2;            // 0..63
  int lseg = (tid & 3) * 8;       // 0,8,16,24
  const _Float16* Ap = &A[(size_t)(m0 + lrow) * K + lseg];
  const _Float16* Bp = &Bt[(size_t)(n0 + lrow) * K + lseg];
  unsigned ldsA = lds_off(&As[lrow][lseg]);
  unsigned ldsB = lds_off(&Bs[lrow][lseg]);
  for (int k0 = 0; k0 < K; k0 += 32) {
    async_b128(ldsA, &Ap[k0]);
    async_b128(ldsB, &Bp[k0]);
    if (k0 + 32 < K) {            // CDNA5 global_prefetch of next tiles
      __builtin_prefetch(&Ap[k0 + 32], 0, 1);
      __builtin_prefetch(&Bp[k0 + 32], 0, 1);
    }
    wait_async0();
    __syncthreads();
    Frag af, b0, b1;
    // A frag: lanes0-15 row=rl, K 0..15; lanes16-31 K 16..31
    af.h[0] = *(const v8h*)&As[mt * 16 + rl][hi * 16];
    af.h[1] = *(const v8h*)&As[mt * 16 + rl][hi * 16 + 8];
    // B frag: lanes0-15 col=rl, K 0..15; lanes16-31 K 16..31
    b0.h[0] = *(const v8h*)&Bs[ntb * 16 + rl][hi * 16];
    b0.h[1] = *(const v8h*)&Bs[ntb * 16 + rl][hi * 16 + 8];
    b1.h[0] = *(const v8h*)&Bs[(ntb + 1) * 16 + rl][hi * 16];
    b1.h[1] = *(const v8h*)&Bs[(ntb + 1) * 16 + rl][hi * 16 + 8];
    acc0 = wmma16x16x32(af.v, b0.v, acc0);
    acc1 = wmma16x16x32(af.v, b1.v, acc1);
    __syncthreads();
  }
  for (int r = 0; r < 8; ++r) {
    int mrow = m0 + mt * 16 + r + 8 * hi;   // VGPR r: M=r (lo lanes) / 8+r (hi)
    int nc0 = n0 + ntb * 16 + rl;
    int nc1 = nc0 + 16;
    float v0 = acc0[r], v1 = acc1[r];
    if (resid) {
      v0 += resid[(size_t)mrow * N + nc0];
      v1 += resid[(size_t)mrow * N + nc1];
    }
    Cout[(size_t)mrow * N + nc0] = v0;
    Cout[(size_t)mrow * N + nc1] = v1;
  }
}

// ---------------------------------------------------------------------------
// 4) RoPE + scale + re-layout to chunk-major [b,h,n,c,dk] f16
// ---------------------------------------------------------------------------
__global__ __launch_bounds__(256) void rope_kernel(const float* __restrict__ src,
                                                   _Float16* __restrict__ dst,
                                                   float scale) {
  int idx = blockIdx.x;         // (b*T+t)*H + h, flattened as row*4 + h
  int h = idx & 3;
  int row = idx >> 2;           // b*T + t
  int t = row & (TT - 1);
  int b = row >> 12;
  int d = threadIdx.x;          // 0..255
  const float* sr = src + (size_t)row * DD + h * DKk;
  float val = sr[d];
  int j = d & 127;
  float theta = (float)t * __powf(10000.f, -(float)j * (1.f / 128.f));
  float c = __cosf(theta), s = __sinf(theta);
  float other = sr[(d < 128) ? d + 128 : d - 128];
  float out = (d < 128) ? (val * c - other * s) : (val * c + other * s);
  out *= scale;
  int n = t >> 6, cc = t & 63;
  size_t o = ((((size_t)(b * HH + h) * NN + n) * CC + cc)) * DKk + d;
  dst[o] = (_Float16)out;
}

// ---------------------------------------------------------------------------
// 5) v: f32 row-major -> chunk-major [b,h,n,c,dv] f16
// ---------------------------------------------------------------------------
__global__ __launch_bounds__(256) void vconvert(const float* __restrict__ src,
                                                _Float16* __restrict__ dst) {
  size_t total = (size_t)ROWS * (2 * DD);
  for (size_t e = blockIdx.x * 256ull + threadIdx.x; e < total;
       e += (size_t)gridDim.x * 256ull) {
    size_t row = e >> 11;             // / 2048
    int col = (int)(e & 2047);
    int h = col >> 9, dvi = col & 511;
    int b = (int)(row >> 12);
    int t = (int)(row & (TT - 1));
    int n = t >> 6, c = t & 63;
    size_t o = ((((size_t)(b * HH + h) * NN + n) * CC + c)) * DVv + dvi;
    dst[o] = (_Float16)src[e];
  }
}

// ---------------------------------------------------------------------------
// 6) Chunked retention. One block per (b, h, dv-tile of 32). 4 waves.
//    S kept transposed in LDS: St[nv][dk] (f16). q/k chunk tiles staged by
//    the Tensor Data Mover (tensor_load_to_lds, TENSORcnt).
// ---------------------------------------------------------------------------
__global__ __launch_bounds__(128) void retention_kernel(
    const _Float16* __restrict__ qh, const _Float16* __restrict__ kh,
    const _Float16* __restrict__ vh, float* __restrict__ o) {
  int dvt = blockIdx.x;    // 0..15 (dv tile of 32)
  int h = blockIdx.y;      // 0..3
  int b = blockIdx.z;      // 0..3
  int tid = threadIdx.x;
  int wave = tid >> 5, lane = tid & 31;
  int rl = lane & 15, hi = lane >> 4;

  __shared__ alignas(32) _Float16 St[32][DKk];    // 16KB state, [nv][d]
  __shared__ alignas(32) _Float16 Qs[CC][128];    // 16KB (half of dk)
  __shared__ alignas(32) _Float16 Ks[CC][128];    // 16KB (also reused as KsT)
  __shared__ alignas(32) _Float16 VsT[32][CC];    // 4KB  [nv][c]
  __shared__ alignas(32) _Float16 ATs[CC][CC];    // 8KB masked attn
  _Float16* KsF = &Ks[0][0];
  unsigned ldsQ = lds_off(&Qs[0][0]);
  unsigned ldsK = lds_off(&Ks[0][0]);

  float gamma = 1.f - exp2f(-5.f - (float)h);
  float lg = log2f(gamma);
  float gC = exp2f(64.f * lg);
  // cross[c] = gamma^(c+1); A-frag rows are constant per lane -> one splat
  _Float16 crossh = (_Float16)exp2f((float)(wave * 16 + rl + 1) * lg);

  for (int e = tid; e < 32 * DKk; e += 128) (&St[0][0])[e] = (_Float16)0.f;
  __syncthreads();

  size_t bh = (size_t)(b * HH + h);
  const _Float16* qb = qh + bh * NN * CC * DKk;
  const _Float16* kb = kh + bh * NN * CC * DKk;
  const _Float16* vb = vh + bh * NN * CC * DVv;

  for (int n = 0; n < NN; ++n) {
    const _Float16* qn = qb + (size_t)n * CC * DKk;
    const _Float16* kn = kb + (size_t)n * CC * DKk;
    const _Float16* vn = vb + (size_t)n * CC * DVv;

    // ---------- Phase A: attn = q @ k^T (64x64, K=256) ----------
    v8f aacc[4] = {{}, {}, {}, {}};   // wave's row tile = wave, all 4 col tiles
    for (int half = 0; half < 2; ++half) {
      if (wave == 0) {   // TDM: 64x128 f16 tile, row stride 256 elems
        tdm_load_2d_f16(ldsQ, qn + half * 128, CC, 128, DKk);
        tdm_load_2d_f16(ldsK, kn + half * 128, CC, 128, DKk);
      }
      if (half == 0) {   // stage v^T tile (32 x 64)
        for (int it = 0; it < 16; ++it) {
          int e = tid + it * 128;
          int nv = e >> 6, c = e & 63;
          VsT[nv][c] = vn[c * DVv + dvt * 32 + nv];
        }
      }
      if (wave == 0) __builtin_amdgcn_s_wait_tensorcnt(0);
      __syncthreads();
      for (int kk = 0; kk < 4; ++kk) {
        Frag af;
        af.h[0] = *(const v8h*)&Qs[wave * 16 + rl][kk * 32 + hi * 16];
        af.h[1] = *(const v8h*)&Qs[wave * 16 + rl][kk * 32 + hi * 16 + 8];
        for (int nt = 0; nt < 4; ++nt) {
          Frag bf;  // B = k^T : rows=d(K), cols=j ; element = Ks[j][d]
          bf.h[0] = *(const v8h*)&Ks[nt * 16 + rl][kk * 32 + hi * 16];
          bf.h[1] = *(const v8h*)&Ks[nt * 16 + rl][kk * 32 + hi * 16 + 8];
          aacc[nt] = wmma16x16x32(af.v, bf.v, aacc[nt]);
        }
      }
      __syncthreads();
    }

    // ---------- Phase B: decay mask, store f16 attn ----------
    for (int nt = 0; nt < 4; ++nt) {
      for (int r = 0; r < 8; ++r) {
        int i = wave * 16 + r + 8 * hi;
        int j = nt * 16 + rl;
        float v = (i >= j) ? aacc[nt][r] * exp2f((float)(i - j) * lg) : 0.f;
        ATs[i][j] = (_Float16)v;
      }
    }
    __syncthreads();

    // ---------- Phase C: o = attn @ v + (q*cross) @ S_old ----------
    v8f oacc[2] = {{}, {}};            // wave's row tile = wave; 2 col tiles
    for (int kk = 0; kk < 2; ++kk) {   // intra-chunk, K = 64
      Frag af;
      af.h[0] = *(const v8h*)&ATs[wave * 16 + rl][kk * 32 + hi * 16];
      af.h[1] = *(const v8h*)&ATs[wave * 16 + rl][kk * 32 + hi * 16 + 8];
      for (int nt = 0; nt < 2; ++nt) {
        Frag bf;  // B = v : rows=c(K), cols=nv ; element = VsT[nv][c]
        bf.h[0] = *(const v8h*)&VsT[nt * 16 + rl][kk * 32 + hi * 16];
        bf.h[1] = *(const v8h*)&VsT[nt * 16 + rl][kk * 32 + hi * 16 + 8];
        oacc[nt] = wmma16x16x32(af.v, bf.v, oacc[nt]);
      }
    }
    for (int half = 0; half < 2; ++half) {  // inter-chunk, K = 256
      if (wave == 0) {  // TDM restage of raw q tile (cross applied in-frag)
        tdm_load_2d_f16(ldsQ, qn + half * 128, CC, 128, DKk);
        __builtin_amdgcn_s_wait_tensorcnt(0);
      }
      __syncthreads();
      for (int kk = 0; kk < 4; ++kk) {
        Frag af;
        af.h[0] = *(const v8h*)&Qs[wave * 16 + rl][kk * 32 + hi * 16];
        af.h[1] = *(const v8h*)&Qs[wave * 16 + rl][kk * 32 + hi * 16 + 8];
        af.v = af.v * crossh;          // packed f16 scale by gamma^(row+1)
        for (int nt = 0; nt < 2; ++nt) {
          Frag bf;  // B = S : rows=d(K), cols=nv ; element = St[nv][d]
          bf.h[0] = *(const v8h*)&St[nt * 16 + rl][half * 128 + kk * 32 + hi * 16];
          bf.h[1] = *(const v8h*)&St[nt * 16 + rl][half * 128 + kk * 32 + hi * 16 + 8];
          oacc[nt] = wmma16x16x32(af.v, bf.v, oacc[nt]);
        }
      }
      __syncthreads();
    }
    // store o (fp32, row-major [b*T+t][H*dv])
    for (int nt = 0; nt < 2; ++nt) {
      for (int r = 0; r < 8; ++r) {
        int i = wave * 16 + r + 8 * hi;
        int t = n * CC + i;
        int col = h * DVv + dvt * 32 + nt * 16 + rl;
        o[((size_t)(b * TT + t)) * (2 * DD) + col] = oacc[nt][r];
      }
    }
    __syncthreads();   // all S_old reads complete before update

    // ---------- Phase D: S = gC*S + (k*kdec)^T @ v ----------
    for (int half = 0; half < 2; ++half) {
      // stage k^T scaled by kdec[c] = gamma^(63-c): KsF[dloc*64 + c]
      for (int it = 0; it < 64; ++it) {
        int e = tid + it * 128;
        int dloc = e >> 6, c = e & 63;
        float kd = exp2f((float)(63 - c) * lg);
        KsF[dloc * 64 + c] =
            (_Float16)((float)kn[c * DKk + half * 128 + dloc] * kd);
      }
      __syncthreads();
      for (int sel = 0; sel < 2; ++sel) {
        int mtl = wave + sel * 4;          // 0..7 (d tile within half)
        for (int nt = 0; nt < 2; ++nt) {
          v8f acc;
          int dbase = half * 128 + mtl * 16 + 8 * hi;
          for (int r = 0; r < 8; ++r)
            acc[r] = gC * (float)St[nt * 16 + rl][dbase + r];
          for (int kk = 0; kk < 2; ++kk) {  // K = c = 64
            Frag af;  // A = (k*kdec)^T : rows=d, K=c ; element = KsF[dloc*64+c]
            const _Float16* ar = &KsF[(mtl * 16 + rl) * 64];
            af.h[0] = *(const v8h*)&ar[kk * 32 + hi * 16];
            af.h[1] = *(const v8h*)&ar[kk * 32 + hi * 16 + 8];
            Frag bf;
            bf.h[0] = *(const v8h*)&VsT[nt * 16 + rl][kk * 32 + hi * 16];
            bf.h[1] = *(const v8h*)&VsT[nt * 16 + rl][kk * 32 + hi * 16 + 8];
            acc = wmma16x16x32(af.v, bf.v, acc);
          }
          for (int r = 0; r < 8; ++r)
            St[nt * 16 + rl][dbase + r] = (_Float16)acc[r];
        }
      }
      __syncthreads();
    }
  }
}

// ---------------------------------------------------------------------------
// 7) RMS-norm + SiLU gate -> f16. One block per (b*T+t, h).
// ---------------------------------------------------------------------------
__global__ __launch_bounds__(128) void gate_kernel(const float* __restrict__ o,
                                                   const float* __restrict__ g,
                                                   const float* __restrict__ gw,
                                                   _Float16* __restrict__ out) {
  int idx = blockIdx.x;
  int h = idx & 3;
  size_t row = (size_t)(idx >> 2);
  int tid = threadIdx.x;
  __shared__ float red[128];
  const float* orow = o + row * (2 * DD) + h * DVv;
  const float* grow = g + row * (2 * DD) + h * DVv;
  float vals[4];
  float s2 = 0.f;
  for (int j = 0; j < 4; ++j) {
    float v = orow[tid + 128 * j];
    vals[j] = v; s2 += v * v;
  }
  red[tid] = s2; __syncthreads();
  for (int off = 64; off; off >>= 1) {
    if (tid < off) red[tid] += red[tid + off];
    __syncthreads();
  }
  float inv = rsqrtf(red[0] * (1.f / DVv) + 1e-5f);
  _Float16* wrow = out + row * (2 * DD) + h * DVv;
  for (int j = 0; j < 4; ++j) {
    int d = tid + 128 * j;
    float gv = grow[d];
    float silu = gv / (1.f + __expf(-gv));
    wrow[d] = (_Float16)(vals[j] * inv * gw[d] * silu);
  }
}

// ---------------------------------------------------------------------------
// Host driver
// ---------------------------------------------------------------------------
extern "C" void kernel_launch(void* const* d_in, const int* in_sizes, int n_in,
                              void* d_out, int out_size, void* d_ws, size_t ws_size,
                              hipStream_t stream) {
  const float* x    = (const float*)d_in[0];
  const float* ln_w = (const float*)d_in[1];
  const float* ln_b = (const float*)d_in[2];
  const float* Wq   = (const float*)d_in[3];
  const float* Wk   = (const float*)d_in[4];
  const float* Wv   = (const float*)d_in[5];
  const float* Wg   = (const float*)d_in[6];
  const float* gnw  = (const float*)d_in[7];
  const float* Wo   = (const float*)d_in[8];
  float* out = (float*)d_out;

  char* ws = (char*)d_ws;
  const size_t MB = 1ull << 20;
  _Float16* hN  = (_Float16*)(ws + 0);         // 32MB normed f16
  _Float16* WqT = (_Float16*)(ws + 32 * MB);   // 2MB
  _Float16* WkT = (_Float16*)(ws + 34 * MB);   // 2MB
  _Float16* WvT = (_Float16*)(ws + 36 * MB);   // 4MB
  _Float16* WgT = (_Float16*)(ws + 40 * MB);   // 4MB
  _Float16* WoT = (_Float16*)(ws + 44 * MB);   // 4MB
  float* q32 = (float*)(ws + 48 * MB);         // 64MB
  float* k32 = (float*)(ws + 112 * MB);        // 64MB
  float* v32 = (float*)(ws + 176 * MB);        // 128MB
  float* g32 = (float*)(ws + 304 * MB);        // 128MB
  _Float16* qh = (_Float16*)(ws + 432 * MB);   // 32MB
  _Float16* kh = (_Float16*)(ws + 464 * MB);   // 32MB
  _Float16* vh = (_Float16*)(ws + 496 * MB);   // 64MB; total 560MB
  float* o32 = q32;          // reuse 128MB (q32+k32 dead after RoPE)
  _Float16* og = qh;         // reuse 64MB (qh+kh dead after retention)

  // 1) LayerNorm
  ln_f16<<<ROWS, 256, 0, stream>>>(x, ln_w, ln_b, hN);

  // 2) weight transpose+convert
  cvtT<<<2048, 256, 0, stream>>>(Wq, WqT, 1024, 1024);
  cvtT<<<2048, 256, 0, stream>>>(Wk, WkT, 1024, 1024);
  cvtT<<<4096, 256, 0, stream>>>(Wv, WvT, 1024, 2048);
  cvtT<<<4096, 256, 0, stream>>>(Wg, WgT, 1024, 2048);
  cvtT<<<4096, 256, 0, stream>>>(Wo, WoT, 2048, 1024);

  // 3) projections (WMMA GEMMs, async-to-LDS staging)
  gemm_f16<<<dim3(16, 256), 256, 0, stream>>>(hN, WqT, nullptr, q32, ROWS, 1024, 1024);
  gemm_f16<<<dim3(16, 256), 256, 0, stream>>>(hN, WkT, nullptr, k32, ROWS, 1024, 1024);
  gemm_f16<<<dim3(32, 256), 256, 0, stream>>>(hN, WvT, nullptr, v32, ROWS, 2048, 1024);
  gemm_f16<<<dim3(32, 256), 256, 0, stream>>>(hN, WgT, nullptr, g32, ROWS, 2048, 1024);

  // 4) RoPE (q scaled by dk^-1/2 = 1/16), chunk-major f16
  rope_kernel<<<ROWS * HH, 256, 0, stream>>>(q32, qh, 1.f / 16.f);
  rope_kernel<<<ROWS * HH, 256, 0, stream>>>(k32, kh, 1.f);

  // 5) v -> chunk-major f16
  vconvert<<<32768, 256, 0, stream>>>(v32, vh);

  // 6) chunked retention (WMMA + TDM)
  retention_kernel<<<dim3(16, HH, BB), 128, 0, stream>>>(qh, kh, vh, o32);

  // 7) RMS-norm + SiLU gate
  gate_kernel<<<ROWS * HH, 128, 0, stream>>>(o32, g32, gnw, og);

  // 8) output projection + residual (WMMA GEMM)
  gemm_f16<<<dim3(16, 256), 256, 0, stream>>>(og, WoT, x, out, ROWS, 1024, 2048);
}
